// InferNet_52261162058294
// MI455X (gfx1250) — matrix operand order repaired and checked by reading, not compiled, not run on hardware
//
#include <hip/hip_runtime.h>
#include <hip/hip_bf16.h>
#include <math.h>

// ---------------------------------------------------------------------------
// Types for CDNA5 WMMA (gfx1250, wave32)
// ---------------------------------------------------------------------------
typedef __bf16 bf16;
typedef __attribute__((ext_vector_type(16))) __bf16 v16bf;
typedef __attribute__((ext_vector_type(8)))  __bf16 v8bf;
typedef __attribute__((ext_vector_type(2)))  __bf16 v2bf;
typedef __attribute__((ext_vector_type(8)))  float  v8f;

#define WMMA_BF16(A_, B_, C_)                                                  \
  __builtin_amdgcn_wmma_f32_16x16x32_bf16(false, (A_), false, (B_),            \
                                          (short)0, (C_), false, false)

#define EPSV 1e-5f

// ---------------------------------------------------------------------------
// Fragment loaders (layouts per CDNA5 ISA 7.12.2)
// ---------------------------------------------------------------------------
// A: 16x32 bf16 tile, row-major [m][k] in LDS (row stride = lda elements).
// Lanes 0-15: M=lane, VGPR v: K = (v/4)*16 + (v%4)*2 (+1); lanes 16-31: +8.
__device__ inline v16bf load_a_frag_lds(const bf16* __restrict__ A, int lda,
                                        int lane) {
  const int m  = lane & 15;
  const int hi = (lane >> 4) & 1;
  const bf16* row = A + m * lda;
  v16bf a;
#pragma unroll
  for (int v = 0; v < 8; ++v) {
    const int k = ((v >> 2) << 4) | (hi << 3) | ((v & 3) << 1);
    a[2 * v]     = row[k];
    a[2 * v + 1] = row[k + 1];
  }
  return a;
}

// B: 32x16 (KxN) bf16 tile, staged TRANSPOSED in LDS as [n][k] (stride ldk).
// Lanes 0-15 hold K=0..7,16..23 of column N=lane; lanes 16-31 hold K=8..15,24..31.
__device__ inline v16bf load_b_frag_lds(const bf16* __restrict__ BT, int ldk,
                                        int lane) {
  const int n  = lane & 15;
  const int k0 = (lane & 16) >> 1;  // 0 or 8
  const bf16* row = BT + n * ldk + k0;
  v16bf b;
#pragma unroll
  for (int j = 0; j < 16; ++j) b[j] = row[(j & 7) + ((j >> 3) << 4)];
  return b;
}

// ---------------------------------------------------------------------------
// Generic WMMA GEMM:  C[M,N] = act( A'[M,K] @ B[K,N] + bias (+ C if ACCUM) )
// A' = A, or row-wise layernorm of A if LN:
//      A'[m,k] = (A[m,k]-mu[m])*rs[m]*lnW[k] + lnB[k]
// Block tile 128(M) x 128(N), K-step 32. 256 threads = 8 waves (2x4),
// each wave owns 4(M) x 2(N) wmma tiles = 8 WMMA per k-step.
// Software-pipelined: next k-step fetched into registers during compute.
// Requires M%128==0, N%128==0, K%32==0.  ACT: 0=none, 1=relu, 2=elu
// ---------------------------------------------------------------------------
template <int ACT, bool LN, bool ACCUM>
__global__ __launch_bounds__(256) void wmma_gemm_kernel(
    const float* __restrict__ A, const float* __restrict__ B,
    float* __restrict__ C, const float* __restrict__ bias,
    const float* __restrict__ rowMu, const float* __restrict__ rowRs,
    const float* __restrict__ lnW, const float* __restrict__ lnB,
    int M, int N, int K) {
  __shared__ bf16 As[128][32];   // 8 KB, [m][k]
  __shared__ bf16 BTs[128][32];  // 8 KB, [n][k]

  const int tid  = threadIdx.x;
  const int lane = tid & 31;
  const int wave = tid >> 5;
  const int wm   = wave >> 2;  // 0..1  (64 rows each)
  const int wn   = wave & 3;   // 0..3  (32 cols each)
  const int bm   = blockIdx.y * 128;
  const int bn   = blockIdx.x * 128;

  v8f acc[4][2] = {};

  // A loader: thread -> (row a_r, 16 consecutive k at a_c)
  const int a_r = tid >> 1;
  const int a_c = (tid & 1) * 16;
  // B loader: thread -> (k row b_k, 16 consecutive n at b_n)
  const int b_k = tid >> 3;
  const int b_n = (tid & 7) * 16;

  float amul = 1.f, aadd = 0.f;
  if constexpr (LN) {
    const float mu = rowMu[bm + a_r];
    const float rs = rowRs[bm + a_r];
    amul = rs;
    aadd = -mu * rs;
  }

  const float* aBase = A + (size_t)(bm + a_r) * K + a_c;
  const float* bBase = B + bn + b_n;

  float4 aReg[4], bReg[4];
  auto loadTiles = [&](int k0) {
    const float4* a4 = reinterpret_cast<const float4*>(aBase + k0);
#pragma unroll
    for (int q = 0; q < 4; ++q) aReg[q] = a4[q];
    const float4* b4 =
        reinterpret_cast<const float4*>(bBase + (size_t)(k0 + b_k) * N);
#pragma unroll
    for (int q = 0; q < 4; ++q) bReg[q] = b4[q];
  };

  loadTiles(0);

  for (int k0 = 0; k0 < K; k0 += 32) {
    // ---- store staged tiles to LDS (f32 -> bf16, LN fused on A) ----
    {
      float va[16];
#pragma unroll
      for (int q = 0; q < 4; ++q) {
        va[4 * q + 0] = aReg[q].x;
        va[4 * q + 1] = aReg[q].y;
        va[4 * q + 2] = aReg[q].z;
        va[4 * q + 3] = aReg[q].w;
      }
      if constexpr (LN) {
        const float4* lw4 = reinterpret_cast<const float4*>(lnW + k0 + a_c);
        const float4* lb4 = reinterpret_cast<const float4*>(lnB + k0 + a_c);
#pragma unroll
        for (int q = 0; q < 4; ++q) {
          const float4 lw = lw4[q];
          const float4 lb = lb4[q];
          va[4 * q + 0] = (va[4 * q + 0] * amul + aadd) * lw.x + lb.x;
          va[4 * q + 1] = (va[4 * q + 1] * amul + aadd) * lw.y + lb.y;
          va[4 * q + 2] = (va[4 * q + 2] * amul + aadd) * lw.z + lb.z;
          va[4 * q + 3] = (va[4 * q + 3] * amul + aadd) * lw.w + lb.w;
        }
      }
      v8bf p0, p1;
#pragma unroll
      for (int j = 0; j < 8; ++j) {
        p0[j] = (bf16)va[j];
        p1[j] = (bf16)va[8 + j];
      }
      *reinterpret_cast<v8bf*>(&As[a_r][a_c])     = p0;  // ds_store_b128
      *reinterpret_cast<v8bf*>(&As[a_r][a_c + 8]) = p1;

      float vb[16];
#pragma unroll
      for (int q = 0; q < 4; ++q) {
        vb[4 * q + 0] = bReg[q].x;
        vb[4 * q + 1] = bReg[q].y;
        vb[4 * q + 2] = bReg[q].z;
        vb[4 * q + 3] = bReg[q].w;
      }
#pragma unroll
      for (int j = 0; j < 16; ++j) BTs[b_n + j][b_k] = (bf16)vb[j];
    }
    __syncthreads();

    // ---- prefetch next k-step into registers (overlaps with compute) ----
    if (k0 + 32 < K) {
      loadTiles(k0 + 32);
      if (k0 + 64 < K)
        __builtin_prefetch(bBase + (size_t)(k0 + 64 + b_k) * N, 0, 1);
    }

    // ---- compute: 8 WMMA per wave ----
    v16bf af[4], bfv[2];
#pragma unroll
    for (int i = 0; i < 4; ++i)
      af[i] = load_a_frag_lds(&As[wm * 64 + i * 16][0], 32, lane);
#pragma unroll
    for (int j = 0; j < 2; ++j)
      bfv[j] = load_b_frag_lds(&BTs[wn * 32 + j * 16][0], 32, lane);
#pragma unroll
    for (int i = 0; i < 4; ++i)
#pragma unroll
      for (int j = 0; j < 2; ++j) acc[i][j] = WMMA_BF16(af[i], bfv[j], acc[i][j]);
    __syncthreads();
  }

  // Epilogue. C/D layout: lane 0-15 -> N=lane,M=r ; lane 16-31 -> M=8+r.
  const int ln_n = lane & 15;
  const int mhi  = (lane >> 4) * 8;
#pragma unroll
  for (int i = 0; i < 4; ++i) {
#pragma unroll
    for (int j = 0; j < 2; ++j) {
      const int nn = bn + wn * 32 + j * 16 + ln_n;
      const float bz = bias ? bias[nn] : 0.f;
#pragma unroll
      for (int r = 0; r < 8; ++r) {
        const int mm = bm + wm * 64 + i * 16 + mhi + r;
        const size_t off = (size_t)mm * N + nn;
        float v = acc[i][j][r] + bz;
        if constexpr (ACCUM) v += C[off];
        if constexpr (ACT == 1) v = fmaxf(v, 0.f);
        if constexpr (ACT == 2) v = v > 0.f ? v : expm1f(v);
        C[off] = v;
      }
    }
  }
}

// ---------------------------------------------------------------------------
// Fused LSTM step (SAGE lstm aggregator). One block = 16 nodes; 256 thr.
// gates[16,1024] = x_t[16,256] @ Wih + h[16,256] @ Whh   (K=512 total)
// Wave w owns col-tiles ct = w + 8*s (s=0..7) so each lane holds matching
// (i,f,g,o) entries for the same (m, j) -> cell update stays in registers.
// ---------------------------------------------------------------------------
__global__ __launch_bounds__(256) void lstm_step_kernel(
    const float* __restrict__ feat,       // [N,F]
    const int* __restrict__ neighbors,    // [N,D]
    const float* __restrict__ Wih,        // [F,4F]
    const float* __restrict__ Whh,        // [F,4F]
    const float* __restrict__ bvec,       // [4F]
    float* __restrict__ hState,           // [N,F]
    float* __restrict__ cState,           // [N,F]
    int t) {
  const int F = 256, D = 16, G = 1024;
  __shared__ bf16 As[16][32];    // 1 KB
  __shared__ bf16 BTs[512][32];  // 32 KB, [n_local][k]

  const int tid   = threadIdx.x;
  const int lane  = tid & 31;
  const int wave  = tid >> 5;
  const int node0 = blockIdx.x * 16;

  v8f acc[8] = {};

  const int kk = tid >> 3;          // B loader: k row 0..31
  const int n0 = (tid & 7) * 64;    // B loader: 64 consecutive cols

  for (int k0 = 0; k0 < 512; k0 += 32) {
    const bool isX = (k0 < 256);
    const float* Bsrc = isX ? Wih : Whh;
    const int kb = isX ? k0 : (k0 - 256);

    __syncthreads();
    {  // stage A: 16x32, 2 elements per thread (float2 -> packed v2bf)
      const int m  = tid >> 4;
      const int kq = (tid & 15) * 2;
      const int node = node0 + m;
      const float* arow;
      if (isX) {
        const int nbr = neighbors[node * D + t];
        arow = feat + (size_t)nbr * F;
      } else {
        arow = hState + (size_t)node * F;
      }
      const float2 xy = *reinterpret_cast<const float2*>(arow + kb + kq);
      v2bf p;
      p[0] = (bf16)xy.x;
      p[1] = (bf16)xy.y;
      *reinterpret_cast<v2bf*>(&As[m][kq]) = p;
    }
    // stage B half 0 (gate cols 0..511): 16x float4 loads, scatter b16 store
    {
      const float4* s4 =
          reinterpret_cast<const float4*>(Bsrc + (size_t)(kb + kk) * G + n0);
#pragma unroll
      for (int q = 0; q < 16; ++q) {
        const float4 y = s4[q];
        BTs[n0 + 4 * q + 0][kk] = (bf16)y.x;
        BTs[n0 + 4 * q + 1][kk] = (bf16)y.y;
        BTs[n0 + 4 * q + 2][kk] = (bf16)y.z;
        BTs[n0 + 4 * q + 3][kk] = (bf16)y.w;
      }
    }
    __syncthreads();

    const v16bf af = load_a_frag_lds(&As[0][0], 32, lane);
#pragma unroll
    for (int s = 0; s < 4; ++s) {
      const int ct = wave + 8 * s;  // 0..31
      const v16bf bfv = load_b_frag_lds(&BTs[ct * 16][0], 32, lane);
      acc[s] = WMMA_BF16(af, bfv, acc[s]);
    }

    __syncthreads();
    // stage B half 1 (gate cols 512..1023)
    {
      const float4* s4 = reinterpret_cast<const float4*>(
          Bsrc + (size_t)(kb + kk) * G + 512 + n0);
#pragma unroll
      for (int q = 0; q < 16; ++q) {
        const float4 y = s4[q];
        BTs[n0 + 4 * q + 0][kk] = (bf16)y.x;
        BTs[n0 + 4 * q + 1][kk] = (bf16)y.y;
        BTs[n0 + 4 * q + 2][kk] = (bf16)y.z;
        BTs[n0 + 4 * q + 3][kk] = (bf16)y.w;
      }
    }
    __syncthreads();
#pragma unroll
    for (int s = 4; s < 8; ++s) {
      const int ct = wave + 8 * s;  // 32..63 -> local (ct*16 - 512)
      const v16bf bfv = load_b_frag_lds(&BTs[ct * 16 - 512][0], 32, lane);
      acc[s] = WMMA_BF16(af, bfv, acc[s]);
    }
  }

  // LSTM cell update, fully in registers: i=acc[s], f=acc[s+2], g=acc[s+4],
  // o=acc[s+6] for s in {0,1}.
  const int nloc = lane & 15;
  const int mhi  = (lane >> 4) * 8;
#pragma unroll
  for (int s = 0; s < 2; ++s) {
    const int jm = (wave + 8 * s) * 16 + nloc;  // gate column 0..255
    const float bi  = bvec[jm];
    const float bf_ = bvec[256 + jm];
    const float bg  = bvec[512 + jm];
    const float bo  = bvec[768 + jm];
#pragma unroll
    for (int r = 0; r < 8; ++r) {
      const int node = node0 + mhi + r;
      const size_t off = (size_t)node * F + jm;
      const float iv = acc[s][r] + bi;
      const float fv = acc[s + 2][r] + bf_;
      const float gv = acc[s + 4][r] + bg;
      const float ov = acc[s + 6][r] + bo;
      const float si = 1.f / (1.f + expf(-iv));
      const float sf = 1.f / (1.f + expf(-fv));
      const float so = 1.f / (1.f + expf(-ov));
      float c = cState[off];
      c = sf * c + si * tanhf(gv);
      cState[off] = c;
      hState[off] = so * tanhf(c);
    }
  }
}

// ---------------------------------------------------------------------------
// Small helper kernels
// ---------------------------------------------------------------------------
__global__ void zero_kernel(float* p, int n) {
  const int i = blockIdx.x * 256 + threadIdx.x;
  if (i < n) p[i] = 0.f;
}

// BatchNorm over axis 0 (per-column), in place. One thread owns one column.
__global__ void bn_col_kernel(float* __restrict__ x, const float* __restrict__ g,
                              const float* __restrict__ b, int rows, int cols) {
  const int c = blockIdx.x * blockDim.x + threadIdx.x;
  if (c >= cols) return;
  float sum = 0.f, sq = 0.f;
  for (int r = 0; r < rows; ++r) {
    const float v = x[(size_t)r * cols + c];
    sum += v;
    sq += v * v;
  }
  const float mu  = sum / rows;
  const float var = sq / rows - mu * mu;
  const float sc  = rsqrtf(var + EPSV) * g[c];
  const float sh  = b[c] - mu * sc;
  for (int r = 0; r < rows; ++r) {
    const size_t o = (size_t)r * cols + c;
    x[o] = x[o] * sc + sh;
  }
}

// Per-row mean / rsqrt(var+eps) for layernorm fusion. One block per row.
__global__ void row_stats_kernel(const float* __restrict__ x, float* __restrict__ mu,
                                 float* __restrict__ rs, int cols) {
  __shared__ float s1[256], s2[256];
  const int r = blockIdx.x;
  float a = 0.f, b = 0.f;
  for (int c = threadIdx.x; c < cols; c += 256) {
    const float v = x[(size_t)r * cols + c];
    a += v;
    b += v * v;
  }
  s1[threadIdx.x] = a;
  s2[threadIdx.x] = b;
  __syncthreads();
  for (int o = 128; o > 0; o >>= 1) {
    if (threadIdx.x < o) {
      s1[threadIdx.x] += s1[threadIdx.x + o];
      s2[threadIdx.x] += s2[threadIdx.x + o];
    }
    __syncthreads();
  }
  if (threadIdx.x == 0) {
    const float m = s1[0] / cols;
    mu[r] = m;
    rs[r] = rsqrtf(s2[0] / cols - m * m + EPSV);
  }
}

// h_neigh[n,f] = mean_t feat[neighbors[n,t], f]   (F == blockDim == 256)
__global__ void mean_agg_kernel(const float* __restrict__ h,
                                const int* __restrict__ nb,
                                float* __restrict__ out) {
  const int n = blockIdx.x;
  const int f = threadIdx.x;
  float s = 0.f;
#pragma unroll
  for (int t = 0; t < 16; ++t) {
    const int j = nb[n * 16 + t];
    s += h[(size_t)j * 256 + f];
  }
  out[(size_t)n * 256 + f] = s * (1.f / 16.f);
}

// z = mean + noise * exp(log_std)
__global__ void reparam_kernel(const float* __restrict__ m, const float* __restrict__ ls,
                               const float* __restrict__ nz, float* __restrict__ z,
                               int n) {
  const int i = blockIdx.x * 256 + threadIdx.x;
  if (i < n) z[i] = m[i] + nz[i] * expf(ls[i]);
}

// FFT self-conv reduces to z @ flip(z)^T.  Build B[k][n] = z[n][(Z-k)%Z].
__global__ void flipT_kernel(const float* __restrict__ z, float* __restrict__ B) {
  const int idx = blockIdx.x * 256 + threadIdx.x;  // over Z*N = 128*4096
  const int k = idx >> 12;
  const int n = idx & 4095;
  const int kk = (128 - k) & 127;
  B[idx] = z[(size_t)n * 128 + kk];
}

// ---------------------------------------------------------------------------
// Host-side orchestration
// ---------------------------------------------------------------------------
extern "C" void kernel_launch(void* const* d_in, const int* in_sizes, int n_in,
                              void* d_out, int out_size, void* d_ws, size_t ws_size,
                              hipStream_t stream) {
  (void)in_sizes; (void)n_in; (void)out_size; (void)ws_size;
  const int N = 4096, D = 16, F = 256, H = 256, Z = 128, G = 1024;

  const int*   neighbors = (const int*)d_in[0];
  const float* noise = (const float*)d_in[1];
  const float* emb   = (const float*)d_in[2];
  const float* Wih   = (const float*)d_in[3];
  const float* Whh   = (const float*)d_in[4];
  const float* lb    = (const float*)d_in[5];
  const float* sWs   = (const float*)d_in[6];
  const float* sWn   = (const float*)d_in[7];
  const float* sb    = (const float*)d_in[8];
  const float* mWs   = (const float*)d_in[9];
  const float* mWn   = (const float*)d_in[10];
  const float* mb    = (const float*)d_in[11];
  const float* bn1g  = (const float*)d_in[12];
  const float* bn1b  = (const float*)d_in[13];
  const float* bn2g  = (const float*)d_in[14];
  const float* bn2b  = (const float*)d_in[15];
  const float* bn3g  = (const float*)d_in[16];
  const float* bn3b  = (const float*)d_in[17];
  const float* lnw   = (const float*)d_in[18];
  const float* lnbv  = (const float*)d_in[19];
  const float* dW    = (const float*)d_in[20];
  const float* db    = (const float*)d_in[21];
  float* out = (float*)d_out;

  float* ws = (float*)d_ws;
  size_t off = 0;
  float* hA    = ws + off; off += (size_t)N * F;
  float* hB    = ws + off; off += (size_t)N * F;
  float* hS    = ws + off; off += (size_t)N * F;   // LSTM h state
  float* cS    = ws + off; off += (size_t)N * F;   // LSTM c state
  float* hNg   = ws + off; off += (size_t)N * H;
  float* meanB = ws + off; off += (size_t)N * Z;
  float* lsB   = ws + off; off += (size_t)N * Z;
  float* zB    = ws + off; off += (size_t)N * Z;
  float* Bsc   = ws + off; off += (size_t)Z * N;
  float* rowMu = ws + off; off += N;
  float* rowRs = ws + off; off += N;
  float* adjA  = ws + off; off += (size_t)N * N;   // 64 MB ping buffer

  const dim3 blk(256);

  // ---- encoder: 3x SAGEConv(lstm aggregator) + ReLU ----
  const float* feat = emb;
  for (int layer = 0; layer < 3; ++layer) {
    zero_kernel<<<dim3((N * F + 255) / 256), blk, 0, stream>>>(hS, N * F);
    zero_kernel<<<dim3((N * F + 255) / 256), blk, 0, stream>>>(cS, N * F);
    for (int t = 0; t < D; ++t)
      lstm_step_kernel<<<dim3(N / 16), blk, 0, stream>>>(
          feat, neighbors, Wih + (size_t)layer * F * G, Whh + (size_t)layer * F * G,
          lb + (size_t)layer * G, hS, cS, t);
    float* hout = (layer == 1) ? hB : hA;
    wmma_gemm_kernel<0, false, false><<<dim3(H / 128, N / 128), blk, 0, stream>>>(
        feat, sWs + (size_t)layer * F * H, hout, nullptr, nullptr, nullptr,
        nullptr, nullptr, N, H, F);
    wmma_gemm_kernel<1, false, true><<<dim3(H / 128, N / 128), blk, 0, stream>>>(
        hS, sWn + (size_t)layer * F * H, hout, sb + (size_t)layer * H, nullptr,
        nullptr, nullptr, nullptr, N, H, F);  // += , bias, relu
    feat = hout;
  }
  // feat == hA here
  bn_col_kernel<<<dim3((H + 255) / 256), blk, 0, stream>>>(hA, bn1g, bn1b, N, H);

  // ---- mean / log_std heads (SAGE mean aggregator) ----
  mean_agg_kernel<<<dim3(N), blk, 0, stream>>>(hA, neighbors, hNg);
  wmma_gemm_kernel<0, false, false><<<dim3(Z / 128, N / 128), blk, 0, stream>>>(
      hA, mWs, meanB, nullptr, nullptr, nullptr, nullptr, nullptr, N, Z, H);
  wmma_gemm_kernel<0, false, true><<<dim3(Z / 128, N / 128), blk, 0, stream>>>(
      hNg, mWn, meanB, mb, nullptr, nullptr, nullptr, nullptr, N, Z, H);
  wmma_gemm_kernel<0, false, false><<<dim3(Z / 128, N / 128), blk, 0, stream>>>(
      hA, mWs + (size_t)H * Z, lsB, nullptr, nullptr, nullptr, nullptr, nullptr,
      N, Z, H);
  wmma_gemm_kernel<0, false, true><<<dim3(Z / 128, N / 128), blk, 0, stream>>>(
      hNg, mWn + (size_t)H * Z, lsB, mb + Z, nullptr, nullptr, nullptr, nullptr,
      N, Z, H);
  reparam_kernel<<<dim3((N * Z + 255) / 256), blk, 0, stream>>>(meanB, lsB, noise,
                                                               zB, N * Z);
  bn_col_kernel<<<dim3((Z + 255) / 256), blk, 0, stream>>>(zB, bn2g, bn2b, N, Z);

  // ---- decoder: FFT self-conv == z @ flip_Z(z)^T (one real GEMM) ----
  flipT_kernel<<<dim3((Z * N) / 256), blk, 0, stream>>>(zB, Bsc);
  wmma_gemm_kernel<0, false, false><<<dim3(N / 128, N / 128), blk, 0, stream>>>(
      zB, Bsc, adjA, nullptr, nullptr, nullptr, nullptr, nullptr, N, N, Z);
  bn_col_kernel<<<dim3(N / 256), blk, 0, stream>>>(adjA, bn3g, bn3b, N, N);

  // ---- DenseNet: 3x elu(layernorm(adj) @ W + b), LN fused into A staging ----
  float* cur = adjA;
  for (int i = 0; i < 3; ++i) {
    float* dst = (i & 1) ? adjA : out;  // L0->out, L1->adjA, L2->out
    row_stats_kernel<<<dim3(N), blk, 0, stream>>>(cur, rowMu, rowRs, N);
    wmma_gemm_kernel<2, true, false><<<dim3(N / 128, N / 128), blk, 0, stream>>>(
        cur, dW + (size_t)i * N * N, dst, db + (size_t)i * N, rowMu, rowRs,
        lnw + (size_t)i * N, lnbv + (size_t)i * N, N, N, N);
    cur = dst;
  }
}
